// UNetSCN3D_81655918231723
// MI455X (gfx1250) — matrix operand order repaired and checked by reading, not compiled, not run on hardware
//
#include <hip/hip_runtime.h>
#include <hip/hip_bf16.h>

// MI455X / gfx1250: 5-conv submanifold sparse CNN stack (27-offset gather +
// 16x16 GEMM per conv) + BatchNorm + ReLU + residuals.
//
// V_WMMA_F32_16X16X32_F16 with TWO rulebook offsets per WMMA (K = 2 x 16 cin).
// Activations are stored in f16 in exactly the A-fragment element order, so
// each lane's gather is ONE 16-byte global_load_b128 with no cvt VALU work.
// Invalid (-1) neighbors are redirected branchlessly to an all-zero row at
// index N, keeping EXEC uniform around the WMMA. Accumulation/BN are f32.

typedef __attribute__((ext_vector_type(16))) _Float16 v16h;
typedef __attribute__((ext_vector_type(8)))  _Float16 v8h;
typedef __attribute__((ext_vector_type(4)))  _Float16 v4h;
typedef __attribute__((ext_vector_type(8)))  float    v8f;

#define KOFF    27
#define NPAIR   14        // ceil(27/2), last pair padded with zeros
#define LAYERS  5

// ---------------------------------------------------------------------------
// Pack weights: W[l][k][cin][cout] (f32) -> per-lane B fragments (f16).
// B layout for v_wmma_f32_16x16x32_f16 (wave32), matching A:
//   lane half h = lane>>4, column N = lane & 15
//   elems 0..7 = cin 8h..8h+7 of offset k0, elems 8..15 = cin 8h..8h+7 of k1
// ---------------------------------------------------------------------------
__global__ void pack_weights_kernel(const float* __restrict__ w_in,
                                    const float* __restrict__ wb,
                                    _Float16* __restrict__ wpk) {
    int tid = blockIdx.x * blockDim.x + threadIdx.x;
    if (tid >= LAYERS * NPAIR * 32) return;
    int lane = tid & 31;
    int p    = (tid >> 5) % NPAIR;
    int l    = tid / (32 * NPAIR);
    const float* W = (l == 0) ? w_in : (wb + (size_t)(l - 1) * KOFF * 256);
    int hh   = lane >> 4;
    int cout = lane & 15;
    _Float16* dst = wpk + ((size_t)(l * NPAIR + p) * 32 + lane) * 16;
    #pragma unroll
    for (int cin = 0; cin < 16; ++cin) {
        int k  = 2 * p + (cin >> 3);
        int ci = 8 * hh + (cin & 7);
        dst[cin] = (k < KOFF) ? (_Float16)W[k * 256 + ci * 16 + cout]
                              : (_Float16)0.f;
    }
}

__global__ void zero_stats_kernel(float* __restrict__ stats) {
    if (threadIdx.x < LAYERS * 64) stats[threadIdx.x] = 0.f;
}

// ---------------------------------------------------------------------------
// Convert f32 features -> f16 activation buffer; zero the sentinel row N.
// Thread i converts 8 elements (one 16B f16 store).
// ---------------------------------------------------------------------------
__global__ void cvt_f16_kernel(const float* __restrict__ src,
                               _Float16* __restrict__ dst,
                               int n8, int N) {
    int i = blockIdx.x * blockDim.x + threadIdx.x;
    if (i >= n8) return;
    float4 u0 = ((const float4*)src)[2 * i];
    float4 u1 = ((const float4*)src)[2 * i + 1];
    v8h h;
    h[0] = (_Float16)u0.x; h[1] = (_Float16)u0.y;
    h[2] = (_Float16)u0.z; h[3] = (_Float16)u0.w;
    h[4] = (_Float16)u1.x; h[5] = (_Float16)u1.y;
    h[6] = (_Float16)u1.z; h[7] = (_Float16)u1.w;
    *(v8h*)(dst + (size_t)i * 8) = h;
    if (i < 2) {                       // sentinel zero row at index N
        v8h z = {};
        *(v8h*)(dst + (size_t)N * 16 + (size_t)i * 8) = z;
    }
}

// ---------------------------------------------------------------------------
// Fused gather + WMMA conv + BN-statistics accumulation.
// One wave = one 16-row tile; block 256 thr = 8 waves = 128 rows.
// A layout (16-bit 16x32, wave32): lane&15 = row M; lane>>4 selects
// cin 0-7 vs 8-15; halves [0..7] = offset k0, halves [8..15] = offset k1.
// ---------------------------------------------------------------------------
__global__ void __launch_bounds__(256)
subm_conv_wmma_kernel(const _Float16* __restrict__ fh,  // [(N+1), 16] f16
                      const int*      __restrict__ nbr, // [27, N]
                      const _Float16* __restrict__ wpk, // this layer's base
                      float* __restrict__ out,          // [N, 16] f32 (pre-BN)
                      float* __restrict__ stats,        // [sum16, sumsq16, ...]
                      int N) {
    __shared__ float lsum[16];
    __shared__ float lsq[16];
    if (threadIdx.x < 16) { lsum[threadIdx.x] = 0.f; lsq[threadIdx.x] = 0.f; }
    __syncthreads();

    const int lane = threadIdx.x & 31;
    const int wv   = threadIdx.x >> 5;
    const int base = (blockIdx.x * 8 + wv) * 16;
    const int row  = base + (lane & 15);
    const int hsel = lane >> 4;                 // 0: cin 0..7, 1: cin 8..15
    const _Float16* fh_off = fh + hsel * 8;

    v8f acc = {0.f, 0.f, 0.f, 0.f, 0.f, 0.f, 0.f, 0.f};

    #pragma unroll
    for (int p = 0; p < NPAIR; ++p) {
        const int k0 = 2 * p;
        const int k1 = 2 * p + 1;
        int idx0 = nbr[(size_t)k0 * N + row];
        int idx1 = (k1 < KOFF) ? nbr[(size_t)k1 * N + row] : -1;
        idx0 = (idx0 < 0) ? N : idx0;           // sentinel zero row
        idx1 = (idx1 < 0) ? N : idx1;

        v8h a0 = *(const v8h*)(fh_off + (size_t)idx0 * 16);
        v8h a1 = *(const v8h*)(fh_off + (size_t)idx1 * 16);
        v16h a = __builtin_shufflevector(a0, a1, 0, 1, 2, 3, 4, 5, 6, 7,
                                                 8, 9, 10, 11, 12, 13, 14, 15);

        v16h b = *(const v16h*)(wpk + ((size_t)p * 32 + lane) * 16);

        acc = __builtin_amdgcn_wmma_f32_16x16x32_f16(
            /*neg_a=*/false, a, /*neg_b=*/false, b,
            /*c_mod=*/(short)0, acc, /*reuse_a=*/false, /*reuse_b=*/false);
    }

    // D layout (f32 16x16): VGPR r -> row base+r (lanes 0-15) / base+8+r
    // (lanes 16-31), column = lane & 15.
    const int col = lane & 15;
    const int r0  = base + hsel * 8;
    #pragma unroll
    for (int r = 0; r < 8; ++r)
        out[(size_t)(r0 + r) * 16 + col] = acc[r];

    // BN statistics: per-lane partial sum / sum-of-squares of channel `col`.
    float s = 0.f, s2 = 0.f;
    #pragma unroll
    for (int r = 0; r < 8; ++r) { float v = acc[r]; s += v; s2 += v * v; }
    atomicAdd(&lsum[col], s);
    atomicAdd(&lsq[col],  s2);
    __syncthreads();
    if (threadIdx.x < 16) {
        atomicAdd(&stats[threadIdx.x],      lsum[threadIdx.x]);
        atomicAdd(&stats[16 + threadIdx.x], lsq[threadIdx.x]);
    }
}

// ---------------------------------------------------------------------------
// (sum, sumsq) -> (scale, shift): scale = g*rsqrt(var+eps),
// shift = b - mean*scale; var = E[x^2]-E[x]^2 (population, like jnp.var).
// ---------------------------------------------------------------------------
__global__ void bn_finalize_kernel(float* __restrict__ stats,
                                   const float* __restrict__ gamma,
                                   const float* __restrict__ beta,
                                   float invN) {
    int c = threadIdx.x;
    if (c < 16) {
        float m   = stats[c] * invN;
        float var = stats[16 + c] * invN - m * m;
        float sc  = gamma[c] * rsqrtf(var + 1.0e-3f);
        stats[32 + c] = sc;
        stats[48 + c] = beta[c] - m * sc;
    }
}

// ---------------------------------------------------------------------------
// dst(f16) = relu(y*scale + shift [+ resid(f16)]);  4 channels per thread.
// Also zeroes the sentinel row N of dst.
// ---------------------------------------------------------------------------
__global__ void bn_relu_h_kernel(const float* __restrict__ y,
                                 const float* __restrict__ stats,  // +32 sc,+48 sh
                                 const _Float16* __restrict__ resid, // nullable
                                 _Float16* __restrict__ dst,
                                 int n4, int N) {
    int i = blockIdx.x * blockDim.x + threadIdx.x;
    if (i >= n4) return;
    const int c0 = (i & 3) * 4;
    float4 yv = ((const float4*)y)[i];
    float ox = yv.x * stats[32 + c0 + 0] + stats[48 + c0 + 0];
    float oy = yv.y * stats[32 + c0 + 1] + stats[48 + c0 + 1];
    float oz = yv.z * stats[32 + c0 + 2] + stats[48 + c0 + 2];
    float ow = yv.w * stats[32 + c0 + 3] + stats[48 + c0 + 3];
    if (resid) {
        v4h rv = *(const v4h*)(resid + (size_t)i * 4);
        ox += (float)rv[0]; oy += (float)rv[1];
        oz += (float)rv[2]; ow += (float)rv[3];
    }
    v4h h;
    h[0] = (_Float16)fmaxf(ox, 0.f); h[1] = (_Float16)fmaxf(oy, 0.f);
    h[2] = (_Float16)fmaxf(oz, 0.f); h[3] = (_Float16)fmaxf(ow, 0.f);
    *(v4h*)(dst + (size_t)i * 4) = h;
    if (i < 4) {                        // sentinel zero row at index N
        v4h z = {};
        *(v4h*)(dst + (size_t)N * 16 + (size_t)i * 4) = z;
    }
}

// ---------------------------------------------------------------------------
// Final: d_out(f32) = relu(y*scale + shift + resid(f16)).
// ---------------------------------------------------------------------------
__global__ void bn_relu_out_kernel(const float* __restrict__ y,
                                   const float* __restrict__ stats,
                                   const _Float16* __restrict__ resid,
                                   float* __restrict__ dst,
                                   int n4) {
    int i = blockIdx.x * blockDim.x + threadIdx.x;
    if (i >= n4) return;
    const int c0 = (i & 3) * 4;
    float4 yv = ((const float4*)y)[i];
    v4h rv = *(const v4h*)(resid + (size_t)i * 4);
    float4 o;
    o.x = fmaxf(yv.x * stats[32 + c0 + 0] + stats[48 + c0 + 0] + (float)rv[0], 0.f);
    o.y = fmaxf(yv.y * stats[32 + c0 + 1] + stats[48 + c0 + 1] + (float)rv[1], 0.f);
    o.z = fmaxf(yv.z * stats[32 + c0 + 2] + stats[48 + c0 + 2] + (float)rv[2], 0.f);
    o.w = fmaxf(yv.w * stats[32 + c0 + 3] + stats[48 + c0 + 3] + (float)rv[3], 0.f);
    ((float4*)dst)[i] = o;
}

// ---------------------------------------------------------------------------
extern "C" void kernel_launch(void* const* d_in, const int* in_sizes, int n_in,
                              void* d_out, int out_size, void* d_ws, size_t ws_size,
                              hipStream_t stream) {
    const float* feats = (const float*)d_in[0];   // [N,16]
    const int*   nbr   = (const int*)  d_in[1];   // [27,N]
    const float* w_in  = (const float*)d_in[2];   // [27,16,16]
    const float* g_in  = (const float*)d_in[3];   // [16]
    const float* b_in  = (const float*)d_in[4];   // [16]
    const float* wb    = (const float*)d_in[5];   // [2,2,27,16,16]
    const float* gs    = (const float*)d_in[6];   // [2,2,16]
    const float* bs    = (const float*)d_in[7];   // [2,2,16]
    float*       outp  = (float*)d_out;

    const int N  = in_sizes[0] / 16;              // 400000 (divisible by 128)
    const float invN = 1.0f / (float)N;
    const size_t NB  = (size_t)N * 16 * sizeof(float);           // f32 buffer
    const size_t NHB = ((size_t)N + 1) * 16 * sizeof(_Float16);  // f16 + sentinel
    const size_t NHB_AL = (NHB + 255) & ~(size_t)255;

    // Workspace: [0,256K) packed f16 weights; [256K,512K) stats (5 x 64 f32:
    // sum16,sumsq16,scale16,shift16); then Y (f32) and three f16 buffers.
    char*     ws    = (char*)d_ws;
    _Float16* wpk   = (_Float16*)ws;
    float*    stats = (float*)(ws + (256u << 10));
    float*    bufY  = (float*)(ws + (512u << 10));
    _Float16* HA    = (_Float16*)(ws + (512u << 10) + NB);
    _Float16* HB    = (_Float16*)(ws + (512u << 10) + NB + NHB_AL);
    _Float16* HC    = (_Float16*)(ws + (512u << 10) + NB + 2 * NHB_AL);

    const size_t LSTRIDE = (size_t)NPAIR * 32 * 16;   // halves per layer

    const int convGrid = N / 128;                 // 8 waves x 16 rows per block
    const int n4       = N * 4;                   // 4-channel groups
    const int n8       = N * 2;                   // 8-element groups
    const int ewGrid   = (n4 + 255) / 256;

    pack_weights_kernel<<<(LAYERS * NPAIR * 32 + 255) / 256, 256, 0, stream>>>(
        w_in, wb, wpk);
    zero_stats_kernel<<<1, LAYERS * 64, 0, stream>>>(stats);
    cvt_f16_kernel<<<(n8 + 255) / 256, 256, 0, stream>>>(feats, HA, n8, N);

    // ---- layer 0: x1 = relu(bn(conv(v))) -> HB
    subm_conv_wmma_kernel<<<convGrid, 256, 0, stream>>>(HA, nbr, wpk,
                                                        bufY, stats, N);
    bn_finalize_kernel<<<1, 16, 0, stream>>>(stats, g_in, b_in, invN);
    bn_relu_h_kernel<<<ewGrid, 256, 0, stream>>>(bufY, stats, nullptr, HB, n4, N);

    // ---- block 0 conv A: t1 = relu(bn(conv(x1))) -> HC
    subm_conv_wmma_kernel<<<convGrid, 256, 0, stream>>>(HB, nbr, wpk + 1 * LSTRIDE,
                                                        bufY, stats + 64, N);
    bn_finalize_kernel<<<1, 16, 0, stream>>>(stats + 64, gs + 0, bs + 0, invN);
    bn_relu_h_kernel<<<ewGrid, 256, 0, stream>>>(bufY, stats + 64, nullptr, HC, n4, N);

    // ---- block 0 conv B: x2 = relu(bn(conv(t1)) + x1) -> HA
    subm_conv_wmma_kernel<<<convGrid, 256, 0, stream>>>(HC, nbr, wpk + 2 * LSTRIDE,
                                                        bufY, stats + 128, N);
    bn_finalize_kernel<<<1, 16, 0, stream>>>(stats + 128, gs + 16, bs + 16, invN);
    bn_relu_h_kernel<<<ewGrid, 256, 0, stream>>>(bufY, stats + 128, HB, HA, n4, N);

    // ---- block 1 conv A: t2 = relu(bn(conv(x2))) -> HC
    subm_conv_wmma_kernel<<<convGrid, 256, 0, stream>>>(HA, nbr, wpk + 3 * LSTRIDE,
                                                        bufY, stats + 192, N);
    bn_finalize_kernel<<<1, 16, 0, stream>>>(stats + 192, gs + 32, bs + 32, invN);
    bn_relu_h_kernel<<<ewGrid, 256, 0, stream>>>(bufY, stats + 192, nullptr, HC, n4, N);

    // ---- block 1 conv B: out = relu(bn(conv(t2)) + x2) -> d_out (f32)
    subm_conv_wmma_kernel<<<convGrid, 256, 0, stream>>>(HC, nbr, wpk + 4 * LSTRIDE,
                                                        bufY, stats + 256, N);
    bn_finalize_kernel<<<1, 16, 0, stream>>>(stats + 256, gs + 48, bs + 48, invN);
    bn_relu_out_kernel<<<ewGrid, 256, 0, stream>>>(bufY, stats + 256, HA, outp, n4);
}
